// BinaryLinear_63471026700466
// MI455X (gfx1250) — matrix operand order repaired and checked by reading, not compiled, not run on hardware
//
#include <hip/hip_runtime.h>
#include <math.h>

typedef __attribute__((ext_vector_type(16))) __bf16 v16bf;
typedef __attribute__((ext_vector_type(4)))  __bf16 v4bf;
typedef __attribute__((ext_vector_type(8)))  float  v8f;

#define TOKENS 8192
#define KDIM   4096
#define NDIM   4096
#define EPSV   1e-5f

#define BM 128
#define BN 128
#define BK 64

// ws layout (floats):
// [0..1023]  per-block partial sums of weight
// [1024]     mean_w
// [1025]     mean_b

// ---------------------------------------------------------------------------
// Kernel 1: per-block partial sums of weight (deterministic tree reductions)
// ---------------------------------------------------------------------------
__global__ __launch_bounds__(256) void wsum_kernel(const float* __restrict__ w,
                                                   float* __restrict__ ws) {
    __shared__ float red[8];
    const int tid = threadIdx.x;
    const float4* w4 = reinterpret_cast<const float4*>(w);
    float s = 0.f;
    for (int q = tid; q < 4096; q += 256) {
        float4 v = w4[(size_t)blockIdx.x * 4096 + q];
        s += (v.x + v.y) + (v.z + v.w);
    }
    for (int m = 16; m >= 1; m >>= 1) s += __shfl_xor(s, m, 32);
    if ((tid & 31) == 0) red[tid >> 5] = s;
    __syncthreads();
    if (tid == 0) {
        float t = 0.f;
        for (int i = 0; i < 8; ++i) t += red[i];
        ws[blockIdx.x] = t;
    }
}

// ---------------------------------------------------------------------------
// Kernel 2: finalize means (weight partials + bias sum), deterministic
// ---------------------------------------------------------------------------
__global__ __launch_bounds__(256) void mean_finalize_kernel(const float* __restrict__ b,
                                                            float* __restrict__ ws) {
    __shared__ float red[16];
    const int tid = threadIdx.x;
    float s = 0.f;
    for (int q = tid; q < 1024; q += 256) s += ws[q];
    float sb = 0.f;
    for (int q = tid; q < 4096; q += 256) sb += b[q];
    for (int m = 16; m >= 1; m >>= 1) {
        s  += __shfl_xor(s,  m, 32);
        sb += __shfl_xor(sb, m, 32);
    }
    if ((tid & 31) == 0) { red[tid >> 5] = s; red[8 + (tid >> 5)] = sb; }
    __syncthreads();
    if (tid == 0) {
        float tw = 0.f, tb = 0.f;
        for (int i = 0; i < 8; ++i) { tw += red[i]; tb += red[8 + i]; }
        ws[1024] = tw * (1.0f / (4096.0f * 4096.0f));   // mean_w
        ws[1025] = tb * (1.0f / 4096.0f);               // mean_b
    }
}

// ---------------------------------------------------------------------------
// Kernel 3: binarized GEMM via v_wmma_f32_16x16x32_bf16
//   grid = (NDIM/BN, TOKENS/BM) = (32, 64), block = 256 threads (8 wave32)
//   Each wave: 1 N-tile (16) x 8 M-tiles; B fragment binarized once per
//   K-step and reused by 16 WMMAs (8 M-tiles x hi/lo split of f32 x).
//   A staged in LDS (bf16 hi/lo, WMMA A-fragment order), double-buffered.
// ---------------------------------------------------------------------------
__global__ __launch_bounds__(256, 1) void bingemm_kernel(const float* __restrict__ x,
                                                         const float* __restrict__ w,
                                                         const float* __restrict__ bias,
                                                         const float* __restrict__ ws,
                                                         float* __restrict__ out) {
    // per buffer: 128 rows x 64 K = 8192 bf16 (16 KB); hi+lo, double-buffered = 64 KB
    __shared__ __bf16 sAhi[2][8192];
    __shared__ __bf16 sAlo[2][8192];

    const int tid  = threadIdx.x;
    const int lane = tid & 31;
    const int wv   = tid >> 5;
    const int m0   = blockIdx.y * BM;
    const int n0   = blockIdx.x * BN + wv * 16;
    const float mean_w = ws[1024];
    const float mean_b = ws[1025];

    const int ncol   = lane & 15;
    const int half   = lane >> 4;
    const int half16 = half << 4;
    const int o      = n0 + ncol;                 // weight row / output column
    const float* wrow = w + (size_t)o * KDIM;

    v8f acc[8];
#pragma unroll
    for (int mt = 0; mt < 8; ++mt) acc[mt] = (v8f){0.f,0.f,0.f,0.f,0.f,0.f,0.f,0.f};

    // ---- staging helpers: 128x64 f32 chunk of x -> bf16 hi/lo in A-frag order.
    // A frag element j on lane l holds row m=l%16, K = j + 8*((j>=8) + l/16).
    // Inverse (K -> hf,j): K<8:hf=0,j=K | 8-15:hf=1,j=K-8 | 16-23:hf=0,j=K-8 | 24-31:hf=1,j=K-16
    float4 pre[8];

    auto load_chunk = [&](int k0) {
#pragma unroll
        for (int p = 0; p < 8; ++p) {
            const int q      = tid + (p << 8);    // 0..2047 float4s
            const int linear = q << 2;
            const int m      = linear >> 6;       // row in slab (0..127)
            const int kk     = linear & 63;       // K within chunk
            pre[p] = *reinterpret_cast<const float4*>(x + (size_t)(m0 + m) * KDIM + k0 + kk);
        }
    };

    auto store_chunk = [&](int bi) {
#pragma unroll
        for (int p = 0; p < 8; ++p) {
            const int q      = tid + (p << 8);
            const int linear = q << 2;
            const int m      = linear >> 6;
            const int kk     = linear & 63;       // multiple of 4 -> same (kbl,hf) group
            const int kbl    = kk >> 5;
            const int kkk    = kk & 31;
            const int hf     = (kkk >> 3) & 1;
            const int j      = (kkk & 7) + ((kkk >> 4) << 3);
            const int mt     = m >> 4;
            const int l      = (hf << 4) + (m & 15);
            const int dst    = ((((kbl << 3) + mt) << 5) | l) * 16 + j;   // j%4==0 -> 8B aligned
            float vv[4] = {pre[p].x, pre[p].y, pre[p].z, pre[p].w};
            v4bf hh, ll;
#pragma unroll
            for (int e = 0; e < 4; ++e) {
                const __bf16 hb = (__bf16)vv[e];
                hh[e] = hb;
                ll[e] = (__bf16)(vv[e] - (float)hb);
            }
            *reinterpret_cast<v4bf*>(&sAhi[bi][dst]) = hh;
            *reinterpret_cast<v4bf*>(&sAlo[bi][dst]) = ll;
        }
    };

    // prologue: stage chunk 0
    load_chunk(0);
    store_chunk(0);
    __syncthreads();

    const int NSTEP = KDIM / BK;   // 64
    for (int step = 0; step < NSTEP; ++step) {
        const int cur = step & 1;
        if (step + 1 < NSTEP) load_chunk((step + 1) * BK);   // global loads overlap compute

        const int kbase = step * BK;
#pragma unroll
        for (int kbl = 0; kbl < 2; ++kbl) {
            // one B fragment per 32-deep K block, reused by 16 WMMAs
            const float* wp = wrow + kbase + (kbl << 5) + (half << 3);
            v16bf bfrag;
#pragma unroll
            for (int g = 0; g < 4; ++g) {
                float4 wq = *reinterpret_cast<const float4*>(wp + 4 * (g & 1) + 16 * (g >> 1));
                bfrag[4 * g + 0] = wq.x > mean_w ? (__bf16)1.0f : (__bf16)0.0f;
                bfrag[4 * g + 1] = wq.y > mean_w ? (__bf16)1.0f : (__bf16)0.0f;
                bfrag[4 * g + 2] = wq.z > mean_w ? (__bf16)1.0f : (__bf16)0.0f;
                bfrag[4 * g + 3] = wq.w > mean_w ? (__bf16)1.0f : (__bf16)0.0f;
            }
#pragma unroll
            for (int mt = 0; mt < 8; ++mt) {
                const int aoff = (((((kbl << 3) + mt) << 5) | lane) << 4);
                v16bf ahi = *reinterpret_cast<const v16bf*>(&sAhi[cur][aoff]);
                v16bf alo = *reinterpret_cast<const v16bf*>(&sAlo[cur][aoff]);
                acc[mt] = __builtin_amdgcn_wmma_f32_16x16x32_bf16(
                    false, ahi, false, bfrag, (short)0, acc[mt], false, false);
                acc[mt] = __builtin_amdgcn_wmma_f32_16x16x32_bf16(
                    false, alo, false, bfrag, (short)0, acc[mt], false, false);
            }
        }
        __syncthreads();
        if (step + 1 < NSTEP) store_chunk((step + 1) & 1);
        __syncthreads();
    }

    // binarized bias + store unnormalized z (LN kernel follows)
    const float be = bias[o] > mean_b ? 1.0f : 0.0f;
#pragma unroll
    for (int mt = 0; mt < 8; ++mt) {
        const int mrow = m0 + (mt << 4) + (half << 3);   // C layout: lanes 16-31 -> M+8
#pragma unroll
        for (int r = 0; r < 8; ++r) {
            out[(size_t)(mrow + r) * NDIM + o] = acc[mt][r] + be;
        }
    }
}

// ---------------------------------------------------------------------------
// Kernel 4: per-row layernorm (biased var) + ReLU, in place. One block per row.
// ---------------------------------------------------------------------------
__global__ __launch_bounds__(256) void lnrelu_kernel(float* __restrict__ out) {
    __shared__ float red[16];
    __shared__ float stats[2];
    const int tid = threadIdx.x;
    float4* row4 = reinterpret_cast<float4*>(out + (size_t)blockIdx.x * NDIM);

    float s = 0.f, s2 = 0.f;
    for (int q = tid; q < 1024; q += 256) {
        float4 v = row4[q];
        s  += (v.x + v.y) + (v.z + v.w);
        s2 += (v.x * v.x + v.y * v.y) + (v.z * v.z + v.w * v.w);
    }
    for (int m = 16; m >= 1; m >>= 1) {
        s  += __shfl_xor(s,  m, 32);
        s2 += __shfl_xor(s2, m, 32);
    }
    if ((tid & 31) == 0) { red[tid >> 5] = s; red[8 + (tid >> 5)] = s2; }
    __syncthreads();
    if (tid == 0) {
        float ts = 0.f, ts2 = 0.f;
        for (int i = 0; i < 8; ++i) { ts += red[i]; ts2 += red[8 + i]; }
        const float mean = ts * (1.0f / 4096.0f);
        float var = ts2 * (1.0f / 4096.0f) - mean * mean;
        var = var < 0.f ? 0.f : var;
        stats[0] = mean;
        stats[1] = 1.0f / (sqrtf(var) + EPSV);
    }
    __syncthreads();
    const float mean = stats[0];
    const float inv  = stats[1];
    for (int q = tid; q < 1024; q += 256) {
        float4 v = row4[q];
        v.x = fmaxf((v.x - mean) * inv, 0.f);
        v.y = fmaxf((v.y - mean) * inv, 0.f);
        v.z = fmaxf((v.z - mean) * inv, 0.f);
        v.w = fmaxf((v.w - mean) * inv, 0.f);
        row4[q] = v;
    }
}

// ---------------------------------------------------------------------------
extern "C" void kernel_launch(void* const* d_in, const int* in_sizes, int n_in,
                              void* d_out, int out_size, void* d_ws, size_t ws_size,
                              hipStream_t stream) {
    const float* x  = (const float*)d_in[0];   // [8192, 4096]
    const float* w  = (const float*)d_in[1];   // [4096, 4096]
    const float* b  = (const float*)d_in[2];   // [4096]
    float* out = (float*)d_out;                // [8192, 4096]
    float* ws  = (float*)d_ws;

    wsum_kernel<<<1024, 256, 0, stream>>>(w, ws);
    mean_finalize_kernel<<<1, 256, 0, stream>>>(b, ws);
    bingemm_kernel<<<dim3(NDIM / BN, TOKENS / BM), 256, 0, stream>>>(x, w, b, ws, out);
    lnrelu_kernel<<<TOKENS, 256, 0, stream>>>(out);
}
